// ImgAttentionBlock_41884521070828
// MI455X (gfx1250) — matrix operand order repaired and checked by reading, not compiled, not run on hardware
//
#include <hip/hip_runtime.h>
#include <hip/hip_bf16.h>

// ---------------------------------------------------------------------------
// Types for CDNA5 WMMA f16 (16x16x32, f32 accumulate)
// ---------------------------------------------------------------------------
typedef __attribute__((ext_vector_type(16))) _Float16 v16h;
typedef __attribute__((ext_vector_type(8)))  _Float16 v8h;
typedef __attribute__((ext_vector_type(8)))  float    v8f;

union H16 { v16h v; v8h h[2]; };

#define WMMA_F32_F16(A_, B_, C_) \
  __builtin_amdgcn_wmma_f32_16x16x32_f16(false, (A_), false, (B_), (short)0, (C_), false, false)

// Problem constants
constexpr int BATCH = 16;
constexpr int NIMG  = 1024;   // 32*32
constexpr int CCH   = 512;
constexpr int LTXT  = 77;
constexpr int LPAD  = 128;    // text rows padded so KV = 1024+128 = 1152
constexpr int NHEAD = 8;
constexpr int HDIM  = 64;
constexpr int KVP   = 1152;   // padded KV length (9 tiles of 128)
constexpr float NEGINF = -1.0e10f;
constexpr float LOG2E  = 1.44269504088896f;

// ---------------------------------------------------------------------------
// 0) f32 -> f16 weight conversion, stored TRANSPOSED [N][K] so GEMM B-tiles
//    stage with coalesced b128 copies (no per-element LDS transpose).
//    Optionally scaled: folds 1/sqrt(HD) into Wq.
// ---------------------------------------------------------------------------
__global__ void conv_wt_kernel(const float* __restrict__ src, _Float16* __restrict__ dst,
                               float scale, int n) {
  int i = blockIdx.x * 256 + threadIdx.x;
  if (i < n) {
    int k = i >> 9, nn = i & 511;          // src is [K=512][N=512] row-major
    dst[nn * 512 + k] = (_Float16)(src[i] * scale);
  }
}

// ---------------------------------------------------------------------------
// 1) GroupNorm stats: one block per (b, group); stats over (N rows, 16 chans)
// ---------------------------------------------------------------------------
__global__ void gn_stats_kernel(const float* __restrict__ x, float* __restrict__ mu,
                                float* __restrict__ rs, int N) {
  int bg = blockIdx.x;
  int b = bg >> 5, g = bg & 31;
  int cnt = N * 16;
  const float* base = x + (size_t)b * N * CCH + g * 16;
  float s = 0.f, s2 = 0.f;
  for (int i = threadIdx.x; i < cnt; i += 256) {
    float v = base[(size_t)(i >> 4) * CCH + (i & 15)];
    s += v; s2 += v * v;
  }
  __shared__ float sh[256], sh2[256];
  sh[threadIdx.x] = s; sh2[threadIdx.x] = s2;
  __syncthreads();
  for (int off = 128; off > 0; off >>= 1) {
    if (threadIdx.x < off) {
      sh[threadIdx.x]  += sh[threadIdx.x + off];
      sh2[threadIdx.x] += sh2[threadIdx.x + off];
    }
    __syncthreads();
  }
  if (threadIdx.x == 0) {
    float inv  = 1.f / (float)cnt;
    float mean = sh[0] * inv;
    float var  = sh2[0] * inv - mean * mean;
    mu[bg] = mean;
    rs[bg] = rsqrtf(var + 1e-6f);
  }
}

// ---------------------------------------------------------------------------
// 2) Apply GN to image: 8 channels/thread, two f16 outputs (q-norm, kv-norm)
// ---------------------------------------------------------------------------
__global__ void gn_apply_img_kernel(const float* __restrict__ x,
                                    const float* __restrict__ mu, const float* __restrict__ rs,
                                    const float* __restrict__ qs, const float* __restrict__ qb,
                                    const float* __restrict__ ks, const float* __restrict__ kb,
                                    _Float16* __restrict__ xq, _Float16* __restrict__ xkv) {
  size_t vi  = (size_t)blockIdx.x * 256 + threadIdx.x;   // 8-wide vector index
  size_t idx = vi * 8;                                   // < 16*1024*512 = 2^23
  int c = (int)(idx & 511);                              // multiple of 8 -> single group
  int b = (int)(idx >> 19);
  int g = c >> 4;
  float m = mu[b * 32 + g], r = rs[b * 32 + g];
  const float4* xp = (const float4*)(x + idx);
  float4 a0 = xp[0], a1 = xp[1];
  float vn[8] = {a0.x, a0.y, a0.z, a0.w, a1.x, a1.y, a1.z, a1.w};
  v8h oq, ok;
#pragma unroll
  for (int j = 0; j < 8; ++j) {
    float v = (vn[j] - m) * r;
    oq[j] = (_Float16)(v * qs[c + j] + qb[c + j]);
    ok[j] = (_Float16)(v * ks[c + j] + kb[c + j]);
  }
  *(v8h*)(xq + idx)  = oq;
  *(v8h*)(xkv + idx) = ok;
}

// 3) Apply GN to text into zero-padded [B, 128, 512] f16 (8 channels/thread)
__global__ void gn_apply_txt_kernel(const float* __restrict__ t,
                                    const float* __restrict__ mu, const float* __restrict__ rs,
                                    const float* __restrict__ ts, const float* __restrict__ tb,
                                    _Float16* __restrict__ out) {
  size_t vi  = (size_t)blockIdx.x * 256 + threadIdx.x;   // < 16*128*512/8 = 2^17
  size_t idx = vi * 8;
  int c   = (int)(idx & 511);
  int row = (int)((idx >> 9) & 127);
  int b   = (int)(idx >> 16);
  v8h o;
  if (row < LTXT) {
    int g = c >> 4;
    float m = mu[b * 32 + g], r = rs[b * 32 + g];
    const float4* tp = (const float4*)(t + ((size_t)b * LTXT + row) * CCH + c);
    float4 a0 = tp[0], a1 = tp[1];
    float vn[8] = {a0.x, a0.y, a0.z, a0.w, a1.x, a1.y, a1.z, a1.w};
#pragma unroll
    for (int j = 0; j < 8; ++j)
      o[j] = (_Float16)((vn[j] - m) * r * ts[c + j] + tb[c + j]);
  } else {
#pragma unroll
    for (int j = 0; j < 8; ++j) o[j] = (_Float16)0.f;
  }
  *(v8h*)(out + idx) = o;
}

// ---------------------------------------------------------------------------
// Shared 128x128x(K=512) WMMA GEMM core.
// A [M,512] f16 row-major; Wt [512,512] f16 PRE-TRANSPOSED [N][K].
// 256 threads = 8 waves (4x2); each wave: 32 rows x 64 cols = 2x4 WMMA tiles.
// Both tiles stage as coalesced b128 copies; B fragments are contiguous 32B.
// ---------------------------------------------------------------------------
__device__ __forceinline__ void gemm_core(const _Float16* __restrict__ A,
                                          const _Float16* __restrict__ Wt,
                                          int rowBase, int n0,
                                          _Float16* Alds, _Float16* Btlds,
                                          v8f acc[2][4]) {
  const int tid  = threadIdx.x;
  const int lane = tid & 31, wave = tid >> 5;
  const int wm = wave >> 1, wn = wave & 1;
  const int lr = lane & 15, half = lane >> 4;
#pragma unroll
  for (int mt = 0; mt < 2; ++mt)
#pragma unroll
    for (int nt = 0; nt < 4; ++nt)
#pragma unroll
      for (int r = 0; r < 8; ++r) acc[mt][nt][r] = 0.f;

  for (int k0 = 0; k0 < 512; k0 += 32) {
#pragma unroll
    for (int i = 0; i < 2; ++i) {
      int ch = tid * 2 + i;                 // 512 chunks each
      int row = ch >> 2, cc = (ch & 3) * 8;
      // A tile [128 x 32]
      *(v8h*)(&Alds[row * 40 + cc]) =
          *(const v8h*)(A + (size_t)(rowBase + row) * 512 + k0 + cc);
      // B tile [128 n x 32 k] from pre-transposed weights
      *(v8h*)(&Btlds[row * 40 + cc]) =
          *(const v8h*)(Wt + (size_t)(n0 + row) * 512 + k0 + cc);
    }
    if (k0 + 32 < 512) {                     // speculative prefetch of next tiles
      __builtin_prefetch(A  + (size_t)(rowBase + (tid >> 1)) * 512 + k0 + 32, 0, 1);
      __builtin_prefetch(Wt + (size_t)(n0 + (tid >> 1)) * 512 + k0 + 32, 0, 1);
    }
    __syncthreads();

    // A fragments: 16 consecutive K starting at 16*half (matches B)
    H16 af[2];
#pragma unroll
    for (int mt = 0; mt < 2; ++mt) {
      const _Float16* ap = &Alds[(wm * 32 + mt * 16 + lr) * 40];
      af[mt].h[0] = *(const v8h*)(ap + 16 * half);
      af[mt].h[1] = *(const v8h*)(ap + 16 * half + 8);
    }
#pragma unroll
    for (int nt = 0; nt < 4; ++nt) {
      // B fragment: column n, 16 consecutive K starting at 16*half (contiguous)
      H16 bf;
      const _Float16* bp = &Btlds[(wn * 64 + nt * 16 + lr) * 40 + 16 * half];
      bf.h[0] = *(const v8h*)bp;
      bf.h[1] = *(const v8h*)(bp + 8);
#pragma unroll
      for (int mt = 0; mt < 2; ++mt)
        acc[mt][nt] = WMMA_F32_F16(af[mt].v, bf.v, acc[mt][nt]);
    }
    __syncthreads();
  }
}

// QKV projection GEMM: scatter into [B, NH, out_seq, 64] f16; for V
// (vt_mode=1) scatter TRANSPOSED into [B, NH, 64, KVP] so attention stages
// V tiles with straight b128 copies.
__global__ void __launch_bounds__(256)
qkv_gemm_kernel(const _Float16* __restrict__ A, const _Float16* __restrict__ Wt,
                _Float16* __restrict__ Out, int seq_shift, int out_seq, int seq_base,
                int vt_mode) {
  __shared__ _Float16 Alds[128 * 40];
  __shared__ _Float16 Btlds[128 * 40];
  v8f acc[2][4];
  int rowBase = blockIdx.x * 128;
  int n0 = blockIdx.y * 128;
  gemm_core(A, Wt, rowBase, n0, Alds, Btlds, acc);

  const int lane = threadIdx.x & 31, wave = threadIdx.x >> 5;
  const int wm = wave >> 1, wn = wave & 1;
  const int lr = lane & 15, half = lane >> 4;
  const int seq_mask = (1 << seq_shift) - 1;
#pragma unroll
  for (int mt = 0; mt < 2; ++mt)
#pragma unroll
    for (int nt = 0; nt < 4; ++nt)
#pragma unroll
      for (int r = 0; r < 8; ++r) {
        int grow = rowBase + wm * 32 + mt * 16 + half * 8 + r;
        int gcol = n0 + wn * 64 + nt * 16 + lr;
        int b = grow >> seq_shift, rib = grow & seq_mask;
        int h = gcol >> 6, dd = gcol & 63;
        size_t o = vt_mode
            ? (((size_t)b * NHEAD + h) * HDIM + dd) * (size_t)out_seq + seq_base + rib
            : (((size_t)b * NHEAD + h) * out_seq + seq_base + rib) * HDIM + dd;
        Out[o] = (_Float16)acc[mt][nt][r];
      }
}

// Output projection GEMM: f32 epilogue + residual add
__global__ void __launch_bounds__(256)
out_proj_kernel(const _Float16* __restrict__ A, const _Float16* __restrict__ Wt,
                const float* __restrict__ X, float* __restrict__ Out) {
  __shared__ _Float16 Alds[128 * 40];
  __shared__ _Float16 Btlds[128 * 40];
  v8f acc[2][4];
  int rowBase = blockIdx.x * 128;
  int n0 = blockIdx.y * 128;
  gemm_core(A, Wt, rowBase, n0, Alds, Btlds, acc);

  const int lane = threadIdx.x & 31, wave = threadIdx.x >> 5;
  const int wm = wave >> 1, wn = wave & 1;
  const int lr = lane & 15, half = lane >> 4;
#pragma unroll
  for (int mt = 0; mt < 2; ++mt)
#pragma unroll
    for (int nt = 0; nt < 4; ++nt)
#pragma unroll
      for (int r = 0; r < 8; ++r) {
        int grow = rowBase + wm * 32 + mt * 16 + half * 8 + r;
        int gcol = n0 + wn * 64 + nt * 16 + lr;
        size_t o = (size_t)grow * 512 + gcol;
        Out[o] = acc[mt][nt][r] + X[o];
      }
}

// ---------------------------------------------------------------------------
// Flash attention: one block = (b, h, 128-query tile); 8 waves x 16 rows.
// KV length 1152 (9 tiles of 128); text/pad columns masked via LDS mask.
// K is [B,NH,KVP,64] row-major; V is pre-transposed [B,NH,64,KVP].
// ---------------------------------------------------------------------------
__global__ void __launch_bounds__(256)
attn_kernel(const _Float16* __restrict__ qg, const _Float16* __restrict__ kg,
            const _Float16* __restrict__ vtg, const int* __restrict__ text_mask,
            _Float16* __restrict__ attn_out) {
  extern __shared__ _Float16 smem[];
  _Float16* Qs = smem;                       // 128 x 72
  _Float16* Ks = Qs + 128 * 72;              // 128 x 72
  _Float16* Vt = Ks + 128 * 72;              // 64 x 136 ([hd][kv])
  _Float16* Ps = Vt + 64 * 136;              // 8 waves x 16 x 136
  unsigned short* maskS = (unsigned short*)(Ps + 8 * 16 * 136);  // 128 entries

  int blk = blockIdx.x;
  int qb = blk & 7, h = (blk >> 3) & 7, b = blk >> 6;
  int tid = threadIdx.x, lane = tid & 31, wave = tid >> 5;
  int lr = lane & 15, half = lane >> 4;

  if (tid < 128)
    maskS[tid] = (tid < LTXT) ? (unsigned short)(text_mask[b * LTXT + tid] != 0)
                              : (unsigned short)0;

  // Load Q tile once
  const _Float16* qbase = qg + ((size_t)(b * NHEAD + h) * NIMG + qb * 128) * HDIM;
#pragma unroll
  for (int i = 0; i < 4; ++i) {
    int ch = tid * 4 + i;
    int row = ch >> 3, cc = (ch & 7) * 8;
    *(v8h*)(&Qs[row * 72 + cc]) = *(const v8h*)(qbase + row * 64 + cc);
  }

  float m_i[8], l_i[8];
#pragma unroll
  for (int r = 0; r < 8; ++r) { m_i[r] = -3.0e38f; l_i[r] = 0.f; }
  v8f Oacc[4];
#pragma unroll
  for (int i = 0; i < 4; ++i)
#pragma unroll
    for (int r = 0; r < 8; ++r) Oacc[i][r] = 0.f;

  const _Float16* kbase  = kg  + (size_t)(b * NHEAD + h) * KVP * HDIM;
  const _Float16* vtbase = vtg + (size_t)(b * NHEAD + h) * HDIM * KVP;
  _Float16* pw = &Ps[wave * 16 * 136];

  __syncthreads();

  // Q A-fragments are loop-invariant: hoist out of the KV loop
  H16 aQ[2];
#pragma unroll
  for (int kk = 0; kk < 2; ++kk) {
    const _Float16* ap = &Qs[(wave * 16 + lr) * 72 + kk * 32];
    aQ[kk].h[0] = *(const v8h*)(ap + 16 * half);
    aQ[kk].h[1] = *(const v8h*)(ap + 16 * half + 8);
  }

  for (int t = 0; t < 9; ++t) {
    // Stage K tile (row-major) and V tile (already transposed) with b128 copies
#pragma unroll
    for (int i = 0; i < 4; ++i) {
      int ch = tid * 4 + i;                  // 1024 chunks each
      int krow = ch >> 3, kcc = (ch & 7) * 8;
      *(v8h*)(&Ks[krow * 72 + kcc]) =
          *(const v8h*)(kbase + (size_t)(t * 128 + krow) * 64 + kcc);
      int vrow = ch >> 4, vcc = (ch & 15) * 8;
      *(v8h*)(&Vt[vrow * 136 + vcc]) =
          *(const v8h*)(vtbase + (size_t)vrow * KVP + t * 128 + vcc);
    }
    __syncthreads();

    // S = Q x K^T : 8 column tiles of 16 kv each
    v8f S[8];
#pragma unroll
    for (int nt = 0; nt < 8; ++nt) {
      v8f c;
#pragma unroll
      for (int r = 0; r < 8; ++r) c[r] = 0.f;
#pragma unroll
      for (int kk = 0; kk < 2; ++kk) {
        H16 bK;  // column = kv index, 16 consecutive hd values (contiguous in Ks)
        const _Float16* bp = &Ks[(nt * 16 + lr) * 72 + kk * 32 + 16 * half];
        bK.h[0] = *(const v8h*)bp;
        bK.h[1] = *(const v8h*)(bp + 8);
        c = WMMA_F32_F16(aQ[kk].v, bK.v, c);
      }
      S[nt] = c;
    }

    // Mask text/pad columns
#pragma unroll
    for (int nt = 0; nt < 8; ++nt) {
      int col = t * 128 + nt * 16 + lr;
      bool valid = (col < NIMG) || (maskS[col - NIMG] != 0);
      if (!valid) {
#pragma unroll
        for (int r = 0; r < 8; ++r) S[nt][r] = NEGINF;
      }
    }

    // Online softmax (rows live in lane-halves; xor masks 1/2/4/8 stay in-half)
    float mnew[8], scl[8], rsum[8];
#pragma unroll
    for (int r = 0; r < 8; ++r) {
      float mx = S[0][r];
#pragma unroll
      for (int nt = 1; nt < 8; ++nt) mx = fmaxf(mx, S[nt][r]);
#pragma unroll
      for (int sm = 1; sm < 16; sm <<= 1) mx = fmaxf(mx, __shfl_xor(mx, sm, 32));
      mnew[r] = fmaxf(m_i[r], mx);
      scl[r]  = exp2f((m_i[r] - mnew[r]) * LOG2E);
      rsum[r] = 0.f;
    }
#pragma unroll
    for (int nt = 0; nt < 8; ++nt)
#pragma unroll
      for (int r = 0; r < 8; ++r) {
        float p = exp2f((S[nt][r] - mnew[r]) * LOG2E);
        S[nt][r] = p;
        rsum[r] += p;
      }
#pragma unroll
    for (int r = 0; r < 8; ++r) {
#pragma unroll
      for (int sm = 1; sm < 16; sm <<= 1) rsum[r] += __shfl_xor(rsum[r], sm, 32);
      l_i[r] = l_i[r] * scl[r] + rsum[r];
      m_i[r] = mnew[r];
    }
#pragma unroll
    for (int i = 0; i < 4; ++i)
#pragma unroll
      for (int r = 0; r < 8; ++r) Oacc[i][r] *= scl[r];

    // Re-layout P (D-fragments -> A-fragments) through this wave's LDS slice.
    // Per-wave region + in-order per-wave LDS: no block barrier needed here.
#pragma unroll
    for (int nt = 0; nt < 8; ++nt)
#pragma unroll
      for (int r = 0; r < 8; ++r)
        pw[(half * 8 + r) * 136 + nt * 16 + lr] = (_Float16)S[nt][r];

    // O += P x V  (kv = 4 chunks of 32, hd = 4 tiles of 16)
#pragma unroll
    for (int kk = 0; kk < 4; ++kk) {
      H16 aP;
      const _Float16* ap = &pw[lr * 136 + kk * 32];
      aP.h[0] = *(const v8h*)(ap + 16 * half);
      aP.h[1] = *(const v8h*)(ap + 16 * half + 8);
#pragma unroll
      for (int nt = 0; nt < 4; ++nt) {
        H16 bV;  // column = hd, 16 consecutive kv (contiguous in Vt)
        const _Float16* bp = &Vt[(nt * 16 + lr) * 136 + kk * 32 + 16 * half];
        bV.h[0] = *(const v8h*)bp;
        bV.h[1] = *(const v8h*)(bp + 8);
        Oacc[nt] = WMMA_F32_F16(aP.v, bV.v, Oacc[nt]);
      }
    }
    __syncthreads();
  }

  // Finalize: O / l_i, store f16 into attn [B, n, NH, HD]
#pragma unroll
  for (int nt = 0; nt < 4; ++nt)
#pragma unroll
    for (int r = 0; r < 8; ++r) {
      int row = qb * 128 + wave * 16 + half * 8 + r;
      float val = Oacc[nt][r] / l_i[r];
      attn_out[(size_t)(b * NIMG + row) * 512 + h * 64 + nt * 16 + lr] = (_Float16)val;
    }
}

// ---------------------------------------------------------------------------
// Host launcher
// ---------------------------------------------------------------------------
extern "C" void kernel_launch(void* const* d_in, const int* in_sizes, int n_in,
                              void* d_out, int out_size, void* d_ws, size_t ws_size,
                              hipStream_t stream) {
  const float* x      = (const float*)d_in[0];
  const float* t_enc  = (const float*)d_in[1];
  const int*   t_mask = (const int*)d_in[2];
  const float* gqs = (const float*)d_in[3];
  const float* gqb = (const float*)d_in[4];
  const float* gks = (const float*)d_in[5];
  const float* gkb = (const float*)d_in[6];
  const float* gts = (const float*)d_in[7];
  const float* gtb = (const float*)d_in[8];
  const float* Wq  = (const float*)d_in[9];
  const float* Wks = (const float*)d_in[10];
  const float* Wvs = (const float*)d_in[11];
  const float* Wkc = (const float*)d_in[12];
  const float* Wvc = (const float*)d_in[13];
  const float* Wo  = (const float*)d_in[14];
  float* out = (float*)d_out;

  char* p = (char*)d_ws;
  auto take = [&](size_t bytes) -> char* {
    char* r = p;
    p += (bytes + 255) & ~(size_t)255;
    return r;
  };
  _Float16* xq_h   = (_Float16*)take((size_t)BATCH * NIMG * CCH * 2);
  _Float16* xkv_h  = (_Float16*)take((size_t)BATCH * NIMG * CCH * 2);
  _Float16* t_h    = (_Float16*)take((size_t)BATCH * LPAD * CCH * 2);
  _Float16* q_h    = (_Float16*)take((size_t)BATCH * NHEAD * NIMG * HDIM * 2);
  _Float16* k_h    = (_Float16*)take((size_t)BATCH * NHEAD * KVP * HDIM * 2);
  _Float16* vt_h   = (_Float16*)take((size_t)BATCH * NHEAD * HDIM * KVP * 2);
  _Float16* attn_h = (_Float16*)take((size_t)BATCH * NIMG * CCH * 2);
  _Float16* wq_t   = (_Float16*)take(512 * 512 * 2);
  _Float16* wks_t  = (_Float16*)take(512 * 512 * 2);
  _Float16* wvs_t  = (_Float16*)take(512 * 512 * 2);
  _Float16* wkc_t  = (_Float16*)take(512 * 512 * 2);
  _Float16* wvc_t  = (_Float16*)take(512 * 512 * 2);
  _Float16* wo_t   = (_Float16*)take(512 * 512 * 2);
  float* mu_i = (float*)take(BATCH * 32 * 4);
  float* rs_i = (float*)take(BATCH * 32 * 4);
  float* mu_t = (float*)take(BATCH * 32 * 4);
  float* rs_t = (float*)take(BATCH * 32 * 4);

  const int WN = 512 * 512;
  // 0) weight conversion + transpose (fold 1/sqrt(64)=0.125 into Wq)
  conv_wt_kernel<<<WN / 256, 256, 0, stream>>>(Wq,  wq_t,  0.125f, WN);
  conv_wt_kernel<<<WN / 256, 256, 0, stream>>>(Wks, wks_t, 1.0f,   WN);
  conv_wt_kernel<<<WN / 256, 256, 0, stream>>>(Wvs, wvs_t, 1.0f,   WN);
  conv_wt_kernel<<<WN / 256, 256, 0, stream>>>(Wkc, wkc_t, 1.0f,   WN);
  conv_wt_kernel<<<WN / 256, 256, 0, stream>>>(Wvc, wvc_t, 1.0f,   WN);
  conv_wt_kernel<<<WN / 256, 256, 0, stream>>>(Wo,  wo_t,  1.0f,   WN);

  // 1) GroupNorm stats
  gn_stats_kernel<<<BATCH * 32, 256, 0, stream>>>(x, mu_i, rs_i, NIMG);
  gn_stats_kernel<<<BATCH * 32, 256, 0, stream>>>(t_enc, mu_t, rs_t, LTXT);

  // 2/3) Apply GN (8 channels per thread)
  gn_apply_img_kernel<<<(BATCH * NIMG * CCH) / (256 * 8), 256, 0, stream>>>(
      x, mu_i, rs_i, gqs, gqb, gks, gkb, xq_h, xkv_h);
  gn_apply_txt_kernel<<<(BATCH * LPAD * CCH) / (256 * 8), 256, 0, stream>>>(
      t_enc, mu_t, rs_t, gts, gtb, t_h);

  // 4) Projections (M=16384 self / M=2048 cross, K=N=512)
  dim3 gSelf(BATCH * NIMG / 128, 4), gCross(BATCH * LPAD / 128, 4);
  qkv_gemm_kernel<<<gSelf, 256, 0, stream>>>(xq_h,  wq_t,  q_h,  10, NIMG, 0, 0);
  qkv_gemm_kernel<<<gSelf, 256, 0, stream>>>(xkv_h, wks_t, k_h,  10, KVP,  0, 0);
  qkv_gemm_kernel<<<gSelf, 256, 0, stream>>>(xkv_h, wvs_t, vt_h, 10, KVP,  0, 1);
  qkv_gemm_kernel<<<gCross, 256, 0, stream>>>(t_h, wkc_t, k_h,  7, KVP, NIMG, 0);
  qkv_gemm_kernel<<<gCross, 256, 0, stream>>>(t_h, wvc_t, vt_h, 7, KVP, NIMG, 1);

  // 5) Flash attention: B*NH*(1024/128) = 1024 blocks
  size_t attn_lds = (size_t)(128 * 72 + 128 * 72 + 64 * 136 + 8 * 16 * 136) * 2 + 256;
  attn_kernel<<<BATCH * NHEAD * (NIMG / 128), 256, attn_lds, stream>>>(
      q_h, k_h, vt_h, t_mask, attn_h);

  // 6) Output projection + residual
  out_proj_kernel<<<gSelf, 256, 0, stream>>>(attn_h, wo_t, x, out);
}
